// BitLinear_84679575208073
// MI455X (gfx1250) — 8-bit matrix operand order repaired and checked by reading, compiled, not run on hardware
//
#include <hip/hip_runtime.h>
#include <stdint.h>


typedef int v4i __attribute__((ext_vector_type(4)));
typedef int v8i __attribute__((ext_vector_type(8)));

#define Q_EPS    1e-5f
#define NORM_EPS 1e-6f

__global__ __launch_bounds__(256) void wabs_partial_kernel(
    const float* __restrict__ w, float* __restrict__ partials, int total4) {
  __shared__ float red[256];
  const int tid = threadIdx.x;
  size_t i = (size_t)blockIdx.x * 256 + tid;
  const size_t stride = (size_t)gridDim.x * 256;
  float s = 0.0f;
  const float4* w4 = (const float4*)w;
  for (; i < (size_t)total4; i += stride) {
    float4 v = w4[i];
    s += fabsf(v.x) + fabsf(v.y) + fabsf(v.z) + fabsf(v.w);
  }
  red[tid] = s;
  __syncthreads();
  for (int off = 128; off > 0; off >>= 1) {
    if (tid < off) red[tid] += red[tid + off];
    __syncthreads();
  }
  if (tid == 0) partials[blockIdx.x] = red[0];
}

__global__ __launch_bounds__(256) void wscale_final_kernel(
    const float* __restrict__ partials, float* __restrict__ scalars,
    int nparts, float inv_nelem) {
  __shared__ float red[256];
  const int tid = threadIdx.x;
  red[tid] = (tid < nparts) ? partials[tid] : 0.0f;
  __syncthreads();
  for (int off = 128; off > 0; off >>= 1) {
    if (tid < off) red[tid] += red[tid + off];
    __syncthreads();
  }
  if (tid == 0) {
    float s = red[0] * inv_nelem;
    scalars[0] = s;
    scalars[1] = 0.5f * (s + Q_EPS);
  }
}

__global__ __launch_bounds__(256) void wquant_kernel(
    const float* __restrict__ w, const float* __restrict__ scalars,
    int8_t* __restrict__ wq, int total4) {
  const float thr = scalars[1];
  size_t i = (size_t)blockIdx.x * 256 + threadIdx.x;
  if (i >= (size_t)total4) return;
  float4 v = ((const float4*)w)[i];
  uint32_t q0 = (fabsf(v.x) > thr) ? ((v.x > 0.0f) ? 1u : 0xFFu) : 0u;
  uint32_t q1 = (fabsf(v.y) > thr) ? ((v.y > 0.0f) ? 1u : 0xFFu) : 0u;
  uint32_t q2 = (fabsf(v.z) > thr) ? ((v.z > 0.0f) ? 1u : 0xFFu) : 0u;
  uint32_t q3 = (fabsf(v.w) > thr) ? ((v.w > 0.0f) ? 1u : 0xFFu) : 0u;
  ((uint32_t*)wq)[i] = q0 | (q1 << 8) | (q2 << 16) | (q3 << 24);
}

__global__ __launch_bounds__(256) void act_quant_kernel(
    const float* __restrict__ x, const float* __restrict__ g,
    int8_t* __restrict__ xq, float* __restrict__ xscale, int K) {
  __shared__ float red[256];
  const int tid = threadIdx.x;
  const size_t row = blockIdx.x;
  const float4* xr = (const float4*)(x + row * (size_t)K);

  float4 xv = xr[tid];
  float ss = xv.x * xv.x + xv.y * xv.y + xv.z * xv.z + xv.w * xv.w;
  red[tid] = ss;
  __syncthreads();
  for (int off = 128; off > 0; off >>= 1) {
    if (tid < off) red[tid] += red[tid + off];
    __syncthreads();
  }
  const float ms = red[0] / (float)K;
  __syncthreads();

  const float inv_rms = rsqrtf(ms + NORM_EPS);
  float4 gv = ((const float4*)g)[tid];
  float4 xn;
  xn.x = xv.x * inv_rms * gv.x;
  xn.y = xv.y * inv_rms * gv.y;
  xn.z = xv.z * inv_rms * gv.z;
  xn.w = xv.w * inv_rms * gv.w;

  float amax = fmaxf(fmaxf(fabsf(xn.x), fabsf(xn.y)),
                     fmaxf(fabsf(xn.z), fabsf(xn.w)));
  red[tid] = amax;
  __syncthreads();
  for (int off = 128; off > 0; off >>= 1) {
    if (tid < off) red[tid] = fmaxf(red[tid], red[tid + off]);
    __syncthreads();
  }
  const float scale = red[0] * (1.0f / 127.0f);
  const float inv_s = 1.0f / (scale + Q_EPS);

  int q0 = (int)fmaxf(-128.0f, fminf(127.0f, rintf(xn.x * inv_s)));
  int q1 = (int)fmaxf(-128.0f, fminf(127.0f, rintf(xn.y * inv_s)));
  int q2 = (int)fmaxf(-128.0f, fminf(127.0f, rintf(xn.z * inv_s)));
  int q3 = (int)fmaxf(-128.0f, fminf(127.0f, rintf(xn.w * inv_s)));
  uint32_t packed = (uint32_t)(q0 & 0xFF) | ((uint32_t)(q1 & 0xFF) << 8) |
                    ((uint32_t)(q2 & 0xFF) << 16) | ((uint32_t)(q3 & 0xFF) << 24);

  const int k     = tid * 4;
  const int chunk = k >> 6;
  const int j     = k & 63;
  const int grp   = j >> 3;
  const int sub   = j & 7;
  const int ngrp  = (grp & 1) + ((grp >> 2) << 1) + (((grp >> 1) & 1) << 2);
  const int soff  = chunk * 64 + ngrp * 8 + sub;
  ((uint32_t*)(xq + row * (size_t)K))[soff >> 2] = packed;

  if (tid == 0) xscale[row] = scale;
}

__device__ __forceinline__ v8i load_frag32(const int8_t* p) {
  v4i f0 = *(const v4i*)(p);
  v4i f1 = *(const v4i*)(p + 16);
  v8i f;
  f[0] = f0.x; f[1] = f0.y; f[2] = f0.z; f[3] = f0.w;
  f[4] = f1.x; f[5] = f1.y; f[6] = f1.z; f[7] = f1.w;
  return f;
}

__device__ __forceinline__ v8i load_b_frag(const int8_t* p) {
  v4i b0 = *(const v4i*)(p);
  v4i b1 = *(const v4i*)(p + 32);
  v8i b;
  b[0] = b0.x; b[1] = b0.y; b[2] = b0.z; b[3] = b0.w;
  b[4] = b1.x; b[5] = b1.y; b[6] = b1.z; b[7] = b1.w;
  return b;
}

__global__ __launch_bounds__(256) void bitlinear_wmma_kernel(
    const int8_t* __restrict__ xq, const int8_t* __restrict__ wq,
    const float* __restrict__ xscale, const float* __restrict__ scalars,
    float* __restrict__ out, int M, int N, int K) {
  const int lane = threadIdx.x & 31;
  const int wave = threadIdx.x >> 5;
  const int l16  = lane & 15;
  const int half = lane >> 4;

  const int m_base = blockIdx.y * 256 + wave * 32;
  const int n_base = blockIdx.x * 64;

  const int8_t* ap0 = xq + (size_t)(m_base + l16) * K + half * 32;
  const int8_t* ap1 = ap0 + (size_t)16 * K;
  const int8_t* bp  = wq + (size_t)(n_base + l16) * K + half * 16;

  v8i acc[8] = {};

  v8i a0c = load_frag32(ap0);
  v8i a1c = load_frag32(ap1);
  v8i bc0 = load_b_frag(bp);
  v8i bc1 = load_b_frag(bp + (size_t)16 * K);
  v8i bc2 = load_b_frag(bp + (size_t)32 * K);
  v8i bc3 = load_b_frag(bp + (size_t)48 * K);

  for (int kk = 0; kk < K - 64; kk += 64) {
    const int kn = kk + 64;
    v8i a0n = load_frag32(ap0 + kn);
    v8i a1n = load_frag32(ap1 + kn);
    v8i bn0 = load_b_frag(bp + kn);
    v8i bn1 = load_b_frag(bp + (size_t)16 * K + kn);
    v8i bn2 = load_b_frag(bp + (size_t)32 * K + kn);
    v8i bn3 = load_b_frag(bp + (size_t)48 * K + kn);

    acc[0] = __builtin_amdgcn_wmma_i32_16x16x64_iu8(true, a0c, true, bc0, acc[0], false, false);
    acc[4] = __builtin_amdgcn_wmma_i32_16x16x64_iu8(true, a1c, true, bc0, acc[4], false, false);
    acc[1] = __builtin_amdgcn_wmma_i32_16x16x64_iu8(true, a0c, true, bc1, acc[1], false, false);
    acc[5] = __builtin_amdgcn_wmma_i32_16x16x64_iu8(true, a1c, true, bc1, acc[5], false, false);
    acc[2] = __builtin_amdgcn_wmma_i32_16x16x64_iu8(true, a0c, true, bc2, acc[2], false, false);
    acc[6] = __builtin_amdgcn_wmma_i32_16x16x64_iu8(true, a1c, true, bc2, acc[6], false, false);
    acc[3] = __builtin_amdgcn_wmma_i32_16x16x64_iu8(true, a0c, true, bc3, acc[3], false, false);
    acc[7] = __builtin_amdgcn_wmma_i32_16x16x64_iu8(true, a1c, true, bc3, acc[7], false, false);

    a0c = a0n; a1c = a1n;
    bc0 = bn0; bc1 = bn1; bc2 = bn2; bc3 = bn3;
  }

  acc[0] = __builtin_amdgcn_wmma_i32_16x16x64_iu8(true, a0c, true, bc0, acc[0], false, false);
  acc[4] = __builtin_amdgcn_wmma_i32_16x16x64_iu8(true, a1c, true, bc0, acc[4], false, false);
  acc[1] = __builtin_amdgcn_wmma_i32_16x16x64_iu8(true, a0c, true, bc1, acc[1], false, false);
  acc[5] = __builtin_amdgcn_wmma_i32_16x16x64_iu8(true, a1c, true, bc1, acc[5], false, false);
  acc[2] = __builtin_amdgcn_wmma_i32_16x16x64_iu8(true, a0c, true, bc2, acc[2], false, false);
  acc[6] = __builtin_amdgcn_wmma_i32_16x16x64_iu8(true, a1c, true, bc2, acc[6], false, false);
  acc[3] = __builtin_amdgcn_wmma_i32_16x16x64_iu8(true, a0c, true, bc3, acc[3], false, false);
  acc[7] = __builtin_amdgcn_wmma_i32_16x16x64_iu8(true, a1c, true, bc3, acc[7], false, false);

  const float wsc = scalars[0];
  float xs0[8], xs1[8];
#pragma unroll
  for (int r = 0; r < 8; ++r) {
    xs0[r] = xscale[m_base + half * 8 + r] * wsc;
    xs1[r] = xscale[m_base + 16 + half * 8 + r] * wsc;
  }

#pragma unroll
  for (int t = 0; t < 4; ++t) {
    const int n = n_base + t * 16 + l16;
#pragma unroll
    for (int r = 0; r < 8; ++r) {
      const int m0 = m_base + half * 8 + r;
      out[(size_t)m0 * N + n] = (float)acc[t][r] * xs0[r];
    }
#pragma unroll
    for (int r = 0; r < 8; ++r) {
      const int m1 = m_base + 16 + half * 8 + r;
      out[(size_t)m1 * N + n] = (float)acc[4 + t][r] * xs1[r];
    }
  }
}

extern "C" void kernel_launch(void* const* d_in, const int* in_sizes, int n_in,
                              void* d_out, int out_size, void* d_ws, size_t ws_size,
                              hipStream_t stream) {
  const float* x      = (const float*)d_in[0];
  const float* weight = (const float*)d_in[1];
  const float* gnorm  = (const float*)d_in[2];

  const int K = in_sizes[2];
  const int N = in_sizes[1] / K;
  const int M = in_sizes[0] / K;

  uint8_t* ws8 = (uint8_t*)d_ws;
  int8_t*  xq       = (int8_t*)ws8;
  int8_t*  wqd      = (int8_t*)(ws8 + (size_t)M * K);
  float*   xscale   = (float*)(ws8 + (size_t)M * K + (size_t)N * K);
  float*   partials = xscale + M;
  float*   scalars  = partials + 256;

  const int wtotal  = N * K;
  const int wtotal4 = wtotal / 4;

  wabs_partial_kernel<<<256, 256, 0, stream>>>(weight, partials, wtotal4);
  wscale_final_kernel<<<1, 256, 0, stream>>>(partials, scalars, 256,
                                             1.0f / (float)wtotal);
  wquant_kernel<<<(wtotal4 + 255) / 256, 256, 0, stream>>>(weight, scalars,
                                                           wqd, wtotal4);
  act_quant_kernel<<<M, 256, 0, stream>>>(x, gnorm, xq, xscale, K);
  dim3 grid(N / 64, M / 256);
  bitlinear_wmma_kernel<<<grid, 256, 0, stream>>>(xq, wqd, xscale, scalars,
                                                  (float*)d_out, M, N, K);
}
